// Toy_MultiHeadAttention_91182155694561
// MI455X (gfx1250) — matrix operand order repaired and checked by reading, not compiled, not run on hardware
//
#include <hip/hip_runtime.h>
#include <hip/hip_bf16.h>

typedef __attribute__((ext_vector_type(16))) _Float16 v16h;
typedef __attribute__((ext_vector_type(8)))  float    v8f;
typedef unsigned int v4u __attribute__((ext_vector_type(4)));
typedef int          v8i __attribute__((ext_vector_type(8)));
typedef int          v4i __attribute__((ext_vector_type(4)));

#define B_  2
#define L_  2048
#define D_  1024
#define H_  16
#define HD_ 64
#define N3_ (3 * D_)   // 3072
#define M_  (B_ * L_)  // 4096

#if defined(__has_builtin)
#if __has_builtin(__builtin_amdgcn_tensor_load_to_lds) && __has_builtin(__builtin_amdgcn_s_wait_tensorcnt)
#define HAVE_TDM 1
#endif
#endif

// ---------------------------------------------------------------------------
// Kernel 1: QKV projection  C[4096,3072] = x[4096,1024] * W[1024,3072]
// ---------------------------------------------------------------------------
__global__ __launch_bounds__(256)
void qkv_gemm(const float* __restrict__ x, const float* __restrict__ w,
              _Float16* __restrict__ Q, _Float16* __restrict__ K,
              _Float16* __restrict__ V)
{
    __shared__ _Float16 At[128][40];  // [m][k], pitch 80B (16B aligned)
    __shared__ _Float16 Bt[128][40];  // [n][k] transposed, pitch 80B

    const int tid  = threadIdx.x;
    const int wave = tid >> 5;
    const int lane = tid & 31;
    const int half = lane >> 4;
    const int lrow = lane & 15;
    const int mw   = wave >> 2;       // 0..1
    const int nw   = wave & 3;        // 0..3
    const int mBlock = blockIdx.y * 128;
    const int nBlock = blockIdx.x * 128;

    v8f acc[4][2] = {};

    for (int k0 = 0; k0 < D_; k0 += 32) {
        // prefetch next K-step tiles into cache hierarchy
        if (k0 + 32 < D_) {
            __builtin_prefetch(x + (size_t)(mBlock + (tid >> 1)) * D_ + k0 + 32 + (tid & 1) * 16, 0, 1);
            __builtin_prefetch(w + (size_t)(k0 + 32 + (tid >> 3)) * N3_ + nBlock + (tid & 7) * 16, 0, 1);
        }
        // ---- stage A tile (f32 -> f16), contiguous b64 stores ----
        #pragma unroll
        for (int i = 0; i < 4; ++i) {
            int idx = tid + i * 256;          // 1024 float4 chunks
            int m   = idx >> 3;               // 8 float4 per row of 32
            int k4  = (idx & 7) * 4;
            float4 a = *(const float4*)(x + (size_t)(mBlock + m) * D_ + k0 + k4);
            _Float16* dst = &At[m][k4];
            dst[0] = (_Float16)a.x; dst[1] = (_Float16)a.y;
            dst[2] = (_Float16)a.z; dst[3] = (_Float16)a.w;
        }
        // ---- stage B tile transposed to [n][k]: pack k-pairs -> b32 stores ----
        #pragma unroll
        for (int i = 0; i < 2; ++i) {
            int idx = tid + i * 256;          // 512 tasks
            int k2  = idx >> 5;               // k-pair 0..15
            int n4  = (idx & 31) * 4;
            const float* p0 = w + (size_t)(k0 + 2 * k2) * N3_ + nBlock + n4;
            float4 b0 = *(const float4*)(p0);
            float4 b1 = *(const float4*)(p0 + N3_);
            #pragma unroll
            for (int j = 0; j < 4; ++j) {
                union { _Float16 h[2]; unsigned u; } pk;
                pk.h[0] = (_Float16)((&b0.x)[j]);
                pk.h[1] = (_Float16)((&b1.x)[j]);
                *(unsigned*)&Bt[n4 + j][2 * k2] = pk.u;
            }
        }
        __syncthreads();

        // ---- A fragments: lane(0-15)=row, K 0..7 & 16..23; lane(16-31): K 8..15 & 24..31
        v16h afrag[4];
        #pragma unroll
        for (int mi = 0; mi < 4; ++mi) {
            const _Float16* ap = &At[mw * 64 + mi * 16 + lrow][half * 8];
            ((uint4*)&afrag[mi])[0] = *(const uint4*)(ap);
            ((uint4*)&afrag[mi])[1] = *(const uint4*)(ap + 16);
        }
        // ---- B fragments: lane holds column n, K 0..7 & 16..23 (half 0) / 8..15 & 24..31 (half 1)
        v16h bfrag[2];
        #pragma unroll
        for (int ni = 0; ni < 2; ++ni) {
            const _Float16* bp = &Bt[nw * 32 + ni * 16 + lrow][half * 8];
            ((uint4*)&bfrag[ni])[0] = *(const uint4*)(bp);
            ((uint4*)&bfrag[ni])[1] = *(const uint4*)(bp + 16);
        }
        #pragma unroll
        for (int mi = 0; mi < 4; ++mi)
            #pragma unroll
            for (int ni = 0; ni < 2; ++ni)
                acc[mi][ni] = __builtin_amdgcn_wmma_f32_16x16x32_f16(
                    false, afrag[mi], false, bfrag[ni],
                    (short)0, acc[mi][ni], false, false);
        __syncthreads();
    }

    // ---- epilogue: scatter into Q/K/V [B][H][L][hd] as f16 ----
    #pragma unroll
    for (int mi = 0; mi < 4; ++mi) {
        #pragma unroll
        for (int ni = 0; ni < 2; ++ni) {
            #pragma unroll
            for (int r = 0; r < 8; ++r) {
                int m = mBlock + mw * 64 + mi * 16 + r + half * 8;
                int n = nBlock + nw * 32 + ni * 16 + lrow;
                int s  = n >> 10;            // 0=q 1=k 2=v
                int hh = (n >> 6) & 15;
                int dd = n & 63;
                int b  = m >> 11;
                int l  = m & (L_ - 1);
                size_t off = (((size_t)b * H_ + hh) * L_ + l) * HD_ + dd;
                _Float16 val = (_Float16)acc[mi][ni][r];
                (s == 0 ? Q : s == 1 ? K : V)[off] = val;
            }
        }
    }
}

// ---------------------------------------------------------------------------
// Kernel 2: causal flash attention. K tile staged by the Tensor Data Mover
// (one tensor_load_to_lds per kv tile, TENSORcnt-tracked); V tile transposed
// manually with packed b32 stores. 4 waves x 16 query rows per workgroup.
// ---------------------------------------------------------------------------
__global__ __launch_bounds__(128)
void attn_kernel(const _Float16* __restrict__ Q, const _Float16* __restrict__ K,
                 const _Float16* __restrict__ V, float* __restrict__ out)
{
    __shared__ _Float16 Kt[32][64];       // [key][hd], TDM-packed (no pitch pad)
    __shared__ _Float16 Vt[64][40];       // [hd][key] transposed, pitch 80B
    __shared__ _Float16 Pl[4][16][40];    // per-wave P staging, pitch 80B

    const int tid  = threadIdx.x;
    const int wave = tid >> 5;
    const int lane = tid & 31;
    const int half = lane >> 4;
    const int lrow = lane & 15;
    const int bh   = blockIdx.y;          // b*H + h
    const int b    = bh >> 4;
    const int h    = bh & 15;
    const int qBlock = blockIdx.x * 64;
    const int q0     = qBlock + wave * 16;

    const _Float16* Qb = Q + (size_t)bh * L_ * HD_;
    const _Float16* Kb = K + (size_t)bh * L_ * HD_;
    const _Float16* Vb = V + (size_t)bh * L_ * HD_;

    // Q fragments for this wave's 16 rows, hd split into two K=32 steps
    v16h qf[2];
    #pragma unroll
    for (int kk = 0; kk < 2; ++kk) {
        const _Float16* qp = Qb + (size_t)(q0 + lrow) * HD_ + kk * 32 + half * 8;
        ((uint4*)&qf[kk])[0] = *(const uint4*)(qp);
        ((uint4*)&qf[kk])[1] = *(const uint4*)(qp + 16);
    }

    v8f o[4] = {};
    float rmax[8], rsum[8];
    #pragma unroll
    for (int r = 0; r < 8; ++r) { rmax[r] = -__builtin_inff(); rsum[r] = 0.f; }
    const float scale = 0.125f;    // 1/sqrt(64)

    const int kvEnd = qBlock + 64;
    for (int kv0 = 0; kv0 < kvEnd; kv0 += 32) {
#if defined(HAVE_TDM)
        // ---- stage K tile [32 keys][64 hd] via Tensor Data Mover ----
        if (wave == 0) {
            unsigned ldsK = (unsigned)(unsigned long long)(void*)&Kt[0][0];
            unsigned long long ga =
                (unsigned long long)(const void*)(Kb + (size_t)kv0 * HD_);
            v4u g0;
            g0[0] = 1u;                                   // count=1, user mode
            g0[1] = ldsK;                                 // lds_addr
            g0[2] = (unsigned)ga;                         // global_addr lo
            g0[3] = (unsigned)((ga >> 32) & 0x1FFFFFFu) | (2u << 30); // hi | type=2
            v8i g1;
            g1[0] = (int)0x00010000u;                     // data_size=2B
            g1[1] = (int)((64u & 0xffffu) << 16);         // tensor_dim0=64 (lo16)
            g1[2] = (int)((2048u & 0xffffu) << 16);       // dim0 hi=0 | tensor_dim1 lo16
            g1[3] = (int)(64u << 16);                     // dim1 hi=0 | tile_dim0=64
            g1[4] = 32;                                   // tile_dim1=32, tile_dim2=0
            g1[5] = 64;                                   // tensor_dim0_stride=64
            g1[6] = 0;
            g1[7] = 0;
            v4i gz4 = {0, 0, 0, 0};
            v8i gz8 = {0, 0, 0, 0, 0, 0, 0, 0};
            __builtin_amdgcn_tensor_load_to_lds(g0, g1, gz4, gz4, gz8, 0);
            __builtin_amdgcn_s_wait_tensorcnt(0);
        }
#else
        // ---- fallback: manual K tile staging ----
        #pragma unroll
        for (int i = 0; i < 2; ++i) {
            int idx = tid + i * 128;
            int key = idx >> 3;
            int c   = (idx & 7) * 8;
            *(uint4*)&Kt[key][c] =
                *(const uint4*)(Kb + (size_t)(kv0 + key) * HD_ + c);
        }
#endif
        // ---- stage V tile transposed [hd][key]: pack key-pairs -> b32 stores ----
        {
            int kp = tid >> 3;            // key pair 0..15
            int c  = (tid & 7) * 8;       // hd chunk
            const _Float16* vp0 = Vb + (size_t)(kv0 + 2 * kp) * HD_ + c;
            uint4 a0 = *(const uint4*)(vp0);
            uint4 a1 = *(const uint4*)(vp0 + HD_);
            const _Float16* h0 = (const _Float16*)&a0;
            const _Float16* h1 = (const _Float16*)&a1;
            #pragma unroll
            for (int j = 0; j < 8; ++j) {
                union { _Float16 h[2]; unsigned u; } pk;
                pk.h[0] = h0[j];
                pk.h[1] = h1[j];
                *(unsigned*)&Vt[c + j][2 * kp] = pk.u;
            }
        }
        __syncthreads();

        // ---- S = Q K^T : two 16-key n-tiles, K-loop over hd ----
        v8f s[2] = {};
        #pragma unroll
        for (int nt = 0; nt < 2; ++nt) {
            #pragma unroll
            for (int kk = 0; kk < 2; ++kk) {
                v16h bf;
                const _Float16* bp = &Kt[nt * 16 + lrow][kk * 32 + half * 8];
                ((uint4*)&bf)[0] = *(const uint4*)(bp);
                ((uint4*)&bf)[1] = *(const uint4*)(bp + 16);
                s[nt] = __builtin_amdgcn_wmma_f32_16x16x32_f16(
                    false, qf[kk], false, bf, (short)0, s[nt], false, false);
            }
        }

        // ---- scale + causal mask ----
        #pragma unroll
        for (int nt = 0; nt < 2; ++nt)
            #pragma unroll
            for (int r = 0; r < 8; ++r) {
                int qq = q0 + r + half * 8;
                int ky = kv0 + nt * 16 + lrow;
                float vv = s[nt][r] * scale;
                s[nt][r] = (ky > qq) ? -__builtin_inff() : vv;
            }

        // ---- online softmax per row ----
        float p[2][8];
        #pragma unroll
        for (int r = 0; r < 8; ++r) {
            float mx = fmaxf(s[0][r], s[1][r]);
            #pragma unroll
            for (int off = 8; off >= 1; off >>= 1)
                mx = fmaxf(mx, __shfl_xor(mx, off, 16));
            float newmax = fmaxf(rmax[r], mx);
            float corr = __expf(rmax[r] - newmax);
            float sum = 0.f;
            #pragma unroll
            for (int nt = 0; nt < 2; ++nt) {
                float e = __expf(s[nt][r] - newmax);
                p[nt][r] = e;
                sum += e;
            }
            #pragma unroll
            for (int off = 8; off >= 1; off >>= 1)
                sum += __shfl_xor(sum, off, 16);
            rsum[r] = rsum[r] * corr + sum;
            rmax[r] = newmax;
            #pragma unroll
            for (int t = 0; t < 4; ++t) o[t][r] *= corr;
        }

        // ---- P (C layout) -> LDS -> A fragment layout ----
        #pragma unroll
        for (int nt = 0; nt < 2; ++nt)
            #pragma unroll
            for (int r = 0; r < 8; ++r)
                Pl[wave][r + half * 8][nt * 16 + lrow] = (_Float16)p[nt][r];

        v16h pf;
        {
            const _Float16* pp = &Pl[wave][lrow][half * 8];
            ((uint4*)&pf)[0] = *(const uint4*)(pp);
            ((uint4*)&pf)[1] = *(const uint4*)(pp + 16);
        }

        // ---- O += P V : 4 hd n-tiles, K=32 keys ----
        #pragma unroll
        for (int t = 0; t < 4; ++t) {
            v16h vf;
            const _Float16* vp2 = &Vt[t * 16 + lrow][half * 8];
            ((uint4*)&vf)[0] = *(const uint4*)(vp2);
            ((uint4*)&vf)[1] = *(const uint4*)(vp2 + 16);
            o[t] = __builtin_amdgcn_wmma_f32_16x16x32_f16(
                false, pf, false, vf, (short)0, o[t], false, false);
        }
        __syncthreads();
    }

    // ---- normalize and write out [B][L][D] f32 ----
    #pragma unroll
    for (int t = 0; t < 4; ++t)
        #pragma unroll
        for (int r = 0; r < 8; ++r) {
            int qq = q0 + r + half * 8;
            int dd = t * 16 + lrow;
            out[((size_t)b * L_ + qq) * D_ + h * HD_ + dd] = o[t][r] / rsum[r];
        }
}

extern "C" void kernel_launch(void* const* d_in, const int* in_sizes, int n_in,
                              void* d_out, int out_size, void* d_ws, size_t ws_size,
                              hipStream_t stream) {
    const float* x = (const float*)d_in[0];
    const float* w = (const float*)d_in[1];

    const size_t perBuf = (size_t)B_ * H_ * L_ * HD_;   // 4M elems
    _Float16* Q = (_Float16*)d_ws;
    _Float16* K = Q + perBuf;
    _Float16* V = K + perBuf;
    float* out = (float*)d_out;

    dim3 g1(N3_ / 128, M_ / 128);       // (24, 32)
    qkv_gemm<<<g1, 256, 0, stream>>>(x, w, Q, K, V);

    dim3 g2(L_ / 64, B_ * H_);          // (32, 32)
    attn_kernel<<<g2, 128, 0, stream>>>(Q, K, V, out);
}
